// APDL_RNN_66013647340234
// MI455X (gfx1250) — matrix operand order repaired and checked by reading, not compiled, not run on hardware
//
#include <hip/hip_runtime.h>

// ---------------------------------------------------------------------------
// APDL-RNN (allpass-delay LSTM scan), MI455X / gfx1250.
// Latency-bound sequential scan: one workgroup (4 waves / 128 threads).
// Wave w owns gate-columns [16w,16w+16) of ALL four gates, so the LSTM
// elementwise combine happens entirely in C-fragment registers -> ONE barrier
// per step (split into s_barrier_signal ... s_barrier_wait; the previous
// step's output stores and the x-term FMAs execute in the window).
// W_hh stays resident in VGPRs as v_wmma_f32_16x16x32_f16 B-fragments.
// bias + x*w_ih is folded into the WMMA C operand; the sigmoid input halving
// is pre-folded into the i/f/o gate weights, so activation inputs come
// straight out of v_wmma into v_tanh_f32. Output stores are forced to the
// saddr + 32-bit-voffset form via inline asm (no 64-bit VALU address chains).
// ---------------------------------------------------------------------------

typedef __attribute__((ext_vector_type(16))) _Float16 v16h;
typedef __attribute__((ext_vector_type(8)))  _Float16 v8h;
typedef __attribute__((ext_vector_type(8)))  float    v8f;

#define B_    16
#define T_    16384
#define H_    64
#define CHUNK 128   // x staging chunk (timesteps); T_ % CHUNK == 0, CHUNK even

// allpass coefficient: OS = 48000/44100, delay_near = floor(OS) = 1
constexpr double OS_    = 48000.0 / 44100.0;
constexpr double ALPHA_ = OS_ - 1.0;
constexpr float  APC    = (float)((1.0 - ALPHA_) / (1.0 + ALPHA_));

// ---- transcendentals: gfx1250 v_tanh_f32 (1 trans op) ----
#if __has_builtin(__builtin_amdgcn_tanhf)
__device__ __forceinline__ float ftanh(float x) { return __builtin_amdgcn_tanhf(x); }
#elif __has_builtin(__builtin_amdgcn_tanh_f32)
__device__ __forceinline__ float ftanh(float x) { return __builtin_amdgcn_tanh_f32(x); }
#else
__device__ __forceinline__ float ftanh(float x) {
    float e = __builtin_amdgcn_exp2f(-2.8853900817779268f * x);   // e^-2x
    return (1.0f - e) * __builtin_amdgcn_rcpf(1.0f + e);
}
#endif

// forced saddr + 32-bit-voffset store (GVS addressing mode)
__device__ __forceinline__ void store_b32(const void* sbase, unsigned boff, float v) {
    asm volatile("global_store_b32 %0, %1, %2"
                 :
                 : "v"(boff), "v"(v), "s"(sbase));
}

__global__ __launch_bounds__(128, 1)
void APDL_RNN_66013647340234_kernel(const float* __restrict__ x,      // (B, T, 1)
                                    const float* __restrict__ W_ih,   // (4H, 1)
                                    const float* __restrict__ W_hh,   // (4H, H)
                                    const float* __restrict__ b_ih,   // (4H)
                                    const float* __restrict__ b_hh,   // (4H)
                                    float* __restrict__ out)          // y | c | ap_final
{
    __shared__ _Float16 h_lds[2][B_][H_];   // published h, double-buffered by parity
    __shared__ float    x_lds[CHUNK][B_];   // transposed input chunk

    const int tid   = threadIdx.x;
    const int lane  = tid & 31;
    const int wave  = tid >> 5;          // wave w owns hidden columns [16w, 16w+16)
    const int nloc  = lane & 15;         // N within the 16x16 tile
    const int mhalf = lane >> 4;         // C-frag: m = r + 8*mhalf
    const int mbase = mhalf * 8;
    const int khalf = mhalf * 8;         // A-frag K sub-offset is 2*khalf (contiguous k order)
    const int kcol  = wave * 16 + nloc;  // owned hidden column (0..63)

    // ---- one-time preload: W_hh columns for all 4 gates as f16 B-fragments ----
    // Gate g tile sits at N-columns 64g + 16w .. +16. B 32x16 f16 K order used here (same as A,
    // not the ISA table order): lanes 0-15 K=0..15, lanes 16-31 K=16..31.
    // i/f/o gates (sigmoid) are pre-scaled by 0.5: sigmoid(z)=0.5*tanh(z/2)+0.5.
    v16h  bfrag[4][2];
    float biasv[4], wihv[4];
#pragma unroll
    for (int g = 0; g < 4; ++g) {
        const float gsc  = (g == 2) ? 1.0f : 0.5f;
        const int   nrow = g * H_ + kcol;        // row of W_hh (gate-major)
#pragma unroll
        for (int kc = 0; kc < 2; ++kc) {
            v16h f;
#pragma unroll
            for (int e = 0; e < 16; ++e) {
                const int k = kc * 32 + ((lane < 16) ? 0 : 16) + e;
                f[e] = (_Float16)(gsc * W_hh[nrow * H_ + k]);
            }
            bfrag[g][kc] = f;
        }
        biasv[g] = gsc * (b_ih[nrow] + b_hh[nrow]);
        wihv[g]  = gsc * W_ih[nrow];   // I == 1
    }

    // recurrent state, held in C-fragment layout: element r <-> (m = mbase+r, kcol)
    v8f ap_h = {0,0,0,0,0,0,0,0}, ap_c = {0,0,0,0,0,0,0,0};
    v8f s1_h = {0,0,0,0,0,0,0,0}, s1_c = {0,0,0,0,0,0,0,0};
    v8f s2_h = {0,0,0,0,0,0,0,0}, s2_c = {0,0,0,0,0,0,0,0};

    const void* youtB = (const void*)out;                              // y base
    const void* coutB = (const void*)(out + (size_t)B_ * T_ * H_);     // c base
    float* __restrict__ apf = out + (size_t)2 * B_ * T_ * H_;

    // 32-bit BYTE offsets per fragment row (shared by the y and c stores)
    unsigned ybase[8];
#pragma unroll
    for (int r = 0; r < 8; ++r)
        ybase[r] = ((unsigned)(mbase + r) * (unsigned)(T_ * H_) + (unsigned)kcol) * 4u;

    // one full recurrence step (par = step parity, compile-time after unroll).
    // sA* = s1 (hc[t-1]) on entry; sB* = s2 (hc[t-2]) on entry, receives hc[t].
    // Caller alternates sA/sB each call -> no history-shift register copies.
    // hc[t-1] (== sA) is stored to global in the barrier window of step t.
    auto step = [&](int t, int ts, int par,
                    v8f& sAh, v8f& sAc, v8f& sBh, v8f& sBc) {
        // x value for this step (latency overlaps allpass + publish)
        v8f xv = *(const v8f*)&x_lds[ts][mbase];

        // ---- allpass state update (registers only) ----
        ap_h = APC * (sAh - ap_h) + sBh;
        ap_c = APC * (sAc - ap_c) + sBc;

        // ---- publish h as f16 (row-major [m][k]) ----
#pragma unroll
        for (int r = 0; r < 8; ++r)
            h_lds[par][mbase + r][kcol] = (_Float16)ap_h[r];

        // ---- split barrier: signal, do independent work, then wait ----
        asm volatile("s_wait_dscnt 0x0\n\ts_barrier_signal -1" ::: "memory");

        // previous step's outputs: hc[t-1] == sA. At t==0 this stores the zero
        // init at t=0, which step 1's store (same wave, same addresses, stores
        // complete in order) overwrites with the real hc[0].
        {
            const int      tp    = (t > 0) ? (t - 1) : 0;            // s_max, uniform
            const unsigned tcolB = (unsigned)tp * (unsigned)(H_ * 4);
#pragma unroll
            for (int r = 0; r < 8; ++r) {
                const unsigned boff = ybase[r] + tcolB;              // 32-bit byte off
                store_b32(youtB, boff, sAh[r]);
                store_b32(coutB, boff, sAc[r]);
            }
        }
        // bias + x*w_ih, used as the WMMA C operand (overlaps barrier/XDL)
        v8f xb[4];
#pragma unroll
        for (int g = 0; g < 4; ++g)
            xb[g] = xv * wihv[g] + biasv[g];

        asm volatile("s_barrier_wait -1" ::: "memory");

        // ---- A fragments (contiguous k = 16*mhalf + s per 32-k chunk, same order as B) ----
        const _Float16* hp = &h_lds[par][nloc][0];
        v8h q0 = *(const v8h*)(hp + 2 * khalf +  0);
        v8h q1 = *(const v8h*)(hp + 2 * khalf +  8);
        v8h q2 = *(const v8h*)(hp + 2 * khalf + 32);
        v8h q3 = *(const v8h*)(hp + 2 * khalf + 40);
        v16h a0 = __builtin_shufflevector(q0, q1, 0,1,2,3,4,5,6,7,8,9,10,11,12,13,14,15);
        v16h a1 = __builtin_shufflevector(q2, q3, 0,1,2,3,4,5,6,7,8,9,10,11,12,13,14,15);

        // ---- 16x16 tiles for all four gates; C preloaded with bias+x term ----
        v8f acc[4];
#pragma unroll
        for (int g = 0; g < 4; ++g) {
            v8f a_ = __builtin_amdgcn_wmma_f32_16x16x32_f16(
                         false, a0, false, bfrag[g][0], (short)0, xb[g], false, false);
            a_     = __builtin_amdgcn_wmma_f32_16x16x32_f16(
                         false, a1, false, bfrag[g][1], (short)0, a_,    false, false);
            acc[g] = a_;
        }

        // ---- activations straight off the WMMA result + in-register combine ----
        v8f hn, cn;
#pragma unroll
        for (int r = 0; r < 8; ++r) {
            const float gi = __builtin_fmaf(0.5f, ftanh(acc[0][r]), 0.5f);
            const float gf = __builtin_fmaf(0.5f, ftanh(acc[1][r]), 0.5f);
            const float gm = ftanh(acc[2][r]);
            const float go = __builtin_fmaf(0.5f, ftanh(acc[3][r]), 0.5f);
            const float cnew = gf * ap_c[r] + gi * gm;
            const float hnew = go * ftanh(cnew);
            cn[r] = cnew;
            hn[r] = hnew;
        }
        sBh = hn;   // becomes s1 for the next step (role swap at call site)
        sBc = cn;
    };

    for (int t0 = 0; t0 < T_; t0 += CHUNK) {
        // ---- stage x[b][t0 .. t0+CHUNK) into LDS, transposed to [t][b] ----
        __syncthreads();   // protect x_lds against readers of previous chunk
        {
            const int sb = tid >> 3;
            const int st = (tid & 7) * (CHUNK / 8);
            const float* xp = x + (size_t)sb * T_ + t0 + st;
#pragma unroll
            for (int j = 0; j < CHUNK / 8; ++j) x_lds[st + j][sb] = xp[j];
        }
        __syncthreads();
        if (t0 + CHUNK < T_)    // global_prefetch_b8 for the next chunk
            __builtin_prefetch(x + (size_t)(tid >> 3) * T_ + (t0 + CHUNK) +
                               (tid & 7) * (CHUNK / 8), 0, 1);

        // 2x unrolled scan: s1/s2 register roles swap, LDS parity is static
        for (int ts = 0; ts < CHUNK; ts += 2) {
            step(t0 + ts,     ts,     0, s1_h, s1_c, s2_h, s2_c);
            step(t0 + ts + 1, ts + 1, 1, s2_h, s2_c, s1_h, s1_c);
        }
    }

    // flush final outputs: after an even number of steps, hc[T-1] lives in s1
    {
        const unsigned tcolB = (unsigned)(T_ - 1) * (unsigned)(H_ * 4);
#pragma unroll
        for (int r = 0; r < 8; ++r) {
            const unsigned boff = ybase[r] + tcolB;
            store_b32(youtB, boff, s1_h[r]);
            store_b32(coutB, boff, s1_c[r]);
        }
    }

    // final allpass carry: ap_final (B, 2H)
#pragma unroll
    for (int r = 0; r < 8; ++r) {
        apf[(mbase + r) * 2 * H_ + kcol]      = ap_h[r];
        apf[(mbase + r) * 2 * H_ + H_ + kcol] = ap_c[r];
    }
}

extern "C" void kernel_launch(void* const* d_in, const int* in_sizes, int n_in,
                              void* d_out, int out_size, void* d_ws, size_t ws_size,
                              hipStream_t stream) {
    (void)in_sizes; (void)n_in; (void)d_ws; (void)ws_size; (void)out_size;
    const float* x    = (const float*)d_in[0];
    const float* W_ih = (const float*)d_in[1];
    const float* W_hh = (const float*)d_in[2];
    const float* b_ih = (const float*)d_in[3];
    const float* b_hh = (const float*)d_in[4];
    float* out = (float*)d_out;

    // Single workgroup: the scan is one serial dependency chain; 4 waves
    // (one per 16-column slice of the hidden state) minimize per-step latency.
    APDL_RNN_66013647340234_kernel<<<dim3(1), dim3(128), 0, stream>>>(
        x, W_ih, W_hh, b_ih, b_hh, out);
}
